// RTNTM_42563125903825
// MI455X (gfx1250) — compile-verified, matrix operand order checked by reading, not run on hardware
//
#include <hip/hip_runtime.h>
#include <math.h>

// Problem constants (from reference setup_inputs)
#define TT 32
#define BB 4
#define DD 256
#define NN 1024
#define MM 256
#define RPL 265      // M + K + 6
#define WPL 775      // 3M + K + 4
#define NROW (TT*BB) // 128 controller rows
#define NCOL (RPL+WPL) // 1040 = 65 * 16, exact tiling
#define NTIL (NCOL/16) // 65
#define EPSV 1e-12f

typedef __attribute__((ext_vector_type(16))) _Float16 v16h;
typedef __attribute__((ext_vector_type(8)))  float    v8f;

// ---------------- workspace layout (bytes) ----------------
static constexpr size_t OFF_BAR  = 0;                                   // barrier counter
static constexpr size_t OFF_WPUB = 256;                                 // [2][B][N] f32
static constexpr size_t OFF_RP   = OFF_WPUB + (size_t)2*BB*NN*4;        // [128][265] f32
static constexpr size_t OFF_WP   = OFF_RP   + (size_t)NROW*RPL*4;       // [128][775] f32
static constexpr size_t OFF_MEM  = OFF_WP   + (size_t)NROW*WPL*4;       // [B][N][M] f32
static constexpr size_t OFF_LINK = OFF_MEM  + (size_t)BB*NN*MM*4;       // [B][N][N] f32

// =====================================================================
// Kernel 1: controller projections for ALL timesteps as one WMMA GEMM.
// P = ctrl(128x256) x Wcat^T(256x1040) + bias, split into rp_all/wp_all.
// One 16x16 output tile per wave; fully unrolled K loop =
// 8 x v_wmma_f32_16x16x32_f16, fragments fetched as float4 (b128) loads.
// =====================================================================
__device__ __forceinline__ v16h load_frag_f16(const float* __restrict__ base,
                                              int k0, int hl)
{
  // 16-bit A/B 16x32 per-lane K mapping: K = (i<8 ? i : i+8) + 8*hl
  // -> two contiguous 8-float runs: [k0+8hl, +8) and [k0+16+8hl, +8)
  const float4 f0 = *(const float4*)(base + k0 + 8*hl + 0);
  const float4 f1 = *(const float4*)(base + k0 + 8*hl + 4);
  const float4 f2 = *(const float4*)(base + k0 + 8*hl + 16);
  const float4 f3 = *(const float4*)(base + k0 + 8*hl + 20);
  v16h v;
  v[0]  = (_Float16)f0.x; v[1]  = (_Float16)f0.y;
  v[2]  = (_Float16)f0.z; v[3]  = (_Float16)f0.w;
  v[4]  = (_Float16)f1.x; v[5]  = (_Float16)f1.y;
  v[6]  = (_Float16)f1.z; v[7]  = (_Float16)f1.w;
  v[8]  = (_Float16)f2.x; v[9]  = (_Float16)f2.y;
  v[10] = (_Float16)f2.z; v[11] = (_Float16)f2.w;
  v[12] = (_Float16)f3.x; v[13] = (_Float16)f3.y;
  v[14] = (_Float16)f3.z; v[15] = (_Float16)f3.w;
  return v;
}

__global__ __launch_bounds__(256) void ntm_proj(
    const float* __restrict__ ctrl,
    const float* __restrict__ rW, const float* __restrict__ rb,
    const float* __restrict__ wW, const float* __restrict__ wb,
    float* __restrict__ rp_all, float* __restrict__ wp_all)
{
  const int nt   = blockIdx.x;           // 0..64 column tile
  const int mt   = threadIdx.x >> 5;     // 0..7  row tile (wave id)
  const int lane = threadIdx.x & 31;
  const int hl   = lane >> 4;            // half-wave selector
  const int l16  = lane & 15;

  const int m = mt*16 + l16;             // A row (always < 128)
  const int n = nt*16 + l16;             // output column (< 1040)

  // B column n = row n of concatenated [read_W ; write_W] (contiguous over K)
  const float* brow = (n < RPL) ? (rW + (size_t)n * DD)
                                : (wW + (size_t)(n - RPL) * DD);
  const float* arow = ctrl + (size_t)m * DD;

  v8f c = {};
  #pragma unroll
  for (int k0 = 0; k0 < DD; k0 += 32) {
    const v16h a   = load_frag_f16(arow, k0, hl);
    const v16h bfr = load_frag_f16(brow, k0, hl);
    c = __builtin_amdgcn_wmma_f32_16x16x32_f16(
            false, a, false, bfr, (short)0, c, false, false);
  }

  #pragma unroll
  for (int v = 0; v < 8; ++v) {
    const int row = mt*16 + v + 8*hl;    // C layout: VGPR v -> M = v (+8 for high half)
    const float val = c[v];
    if (n < RPL) rp_all[(size_t)row*RPL + n]          = val + rb[n];
    else         wp_all[(size_t)row*WPL + (n - RPL)]  = val + wb[n - RPL];
  }
}

// ---------------- block reduction helpers (wave32) ----------------
__device__ __forceinline__ float warp_sum(float v) {
  #pragma unroll
  for (int o = 16; o > 0; o >>= 1) v += __shfl_xor(v, o);
  return v;
}
__device__ __forceinline__ float warp_max(float v) {
  #pragma unroll
  for (int o = 16; o > 0; o >>= 1) v = fmaxf(v, __shfl_xor(v, o));
  return v;
}
__device__ float block_sum(float v, float* s_red) {
  v = warp_sum(v);
  if ((threadIdx.x & 31) == 0) s_red[threadIdx.x >> 5] = v;
  __syncthreads();
  if (threadIdx.x < 32) {
    float r = warp_sum(s_red[threadIdx.x]);
    if (threadIdx.x == 0) s_red[0] = r;
  }
  __syncthreads();
  float r = s_red[0];
  __syncthreads();
  return r;
}
__device__ float block_max(float v, float* s_red) {
  v = warp_max(v);
  if ((threadIdx.x & 31) == 0) s_red[threadIdx.x >> 5] = v;
  __syncthreads();
  if (threadIdx.x < 32) {
    float r = warp_max(s_red[threadIdx.x]);
    if (threadIdx.x == 0) s_red[0] = r;
  }
  __syncthreads();
  float r = s_red[0];
  __syncthreads();
  return r;
}

__device__ __forceinline__ float softplusf(float x) {
  return (x > 20.f) ? x : log1pf(expf(x));
}
__device__ __forceinline__ float sigmoidf(float x) {
  return 1.f / (1.f + expf(-x));
}

// Grid barrier across the 4 persistent blocks (monotonic counter, s_sleep spin)
__device__ void gbar(unsigned int* cnt, unsigned int target) {
  __syncthreads();
  __threadfence();
  if (threadIdx.x == 0) {
    atomicAdd(cnt, 1u);
    while (atomicAdd(cnt, 0u) < target) __builtin_amdgcn_s_sleep(2);
  }
  __syncthreads();
}

// Content + gate + shift + sharpen addressing. Thread n owns memory row n.
// Writes result into s_out[N]; s_wg is scratch.
__device__ void addressing(const float* __restrict__ prm,      // LDS params (key@0, beta@M, gate@M+1, shift@M+2..M+4, gamma@M+5)
                           const float* __restrict__ s_prev,   // LDS [N] previous weights
                           const float* __restrict__ membase,  // global [N][M]
                           float* s_kn, float* s_wg, float* s_out, float* s_red)
{
  const int tid = threadIdx.x;

  // normalized key
  const float kv  = (tid < MM) ? prm[tid] : 0.f;
  const float ksq = block_sum(kv * kv, s_red);
  if (tid < MM) s_kn[tid] = kv / (sqrtf(ksq) + EPSV);
  __syncthreads();

  // cosine similarity of row tid vs key (vectorized 128-bit loads)
  const float4* row = (const float4*)(membase + (size_t)tid * MM);
  const float4* kn4 = (const float4*)s_kn;
  float dot = 0.f, ss = 0.f;
  #pragma unroll 4
  for (int i = 0; i < MM/4; ++i) {
    const float4 mv = row[i], kk = kn4[i];
    dot += mv.x*kk.x + mv.y*kk.y + mv.z*kk.z + mv.w*kk.w;
    ss  += mv.x*mv.x + mv.y*mv.y + mv.z*mv.z + mv.w*mv.w;
  }
  const float cosv   = dot / (sqrtf(ss) + EPSV);
  const float beta_p = softplusf(prm[MM]);
  const float x  = beta_p * cosv;

  // softmax over N
  const float mx = block_max(x, s_red);
  const float e  = expf(x - mx);
  const float es = block_sum(e, s_red);
  const float wc = e / es;

  // interpolation gate
  const float g  = sigmoidf(prm[MM + 1]);
  const float wg = g * wc + (1.f - g) * s_prev[tid];
  s_wg[tid] = wg;
  __syncthreads();

  // 3-tap circular shift (softmax over shift logits, computed per-thread)
  const float sh0 = prm[MM + 2], sh1 = prm[MM + 3], sh2 = prm[MM + 4];
  const float m3  = fmaxf(sh0, fmaxf(sh1, sh2));
  const float e0 = expf(sh0 - m3), e1 = expf(sh1 - m3), e2 = expf(sh2 - m3);
  const float inv = 1.f / (e0 + e1 + e2);
  const float shifted = inv * (e0 * s_wg[(tid + 1) & (NN - 1)]
                             + e1 * wg
                             + e2 * s_wg[(tid + NN - 1) & (NN - 1)]);

  // sharpen + renormalize
  const float gam_p = 1.f + softplusf(prm[MM + 5]);
  const float wt  = powf(shifted + EPSV, gam_p);
  const float wts = block_sum(wt, s_red);
  s_out[tid] = wt / (wts + EPSV);
  __syncthreads();
}

// =====================================================================
// Kernel 2: persistent recurrent kernel. One block per batch (1024 thr),
// state in LDS + L2-resident workspace, grid barrier for the cross-batch
// precedence coupling (reference sums w over the batch axis).
// NOTE: reference _allocate() returns exactly 0 (cumprod of a zero-led
// vector), so w = (1 - a_gate) * w_addr and the usage sort is elided.
// =====================================================================
__global__ __launch_bounds__(1024) void ntm_step(
    const float* __restrict__ mem0,  const float* __restrict__ link0,
    const float* __restrict__ prec0, const float* __restrict__ usage0,
    const float* __restrict__ rw0,   const float* __restrict__ ww0,
    const float* __restrict__ rp_all,const float* __restrict__ wp_all,
    float* __restrict__ mem, float* __restrict__ link,
    float* __restrict__ wpub, unsigned int* __restrict__ bar,
    float* __restrict__ out)
{
  const int b   = blockIdx.x;
  const int tid = threadIdx.x;

  __shared__ float s_rp[RPL + 7];
  __shared__ float s_wp[WPL + 1];
  __shared__ float s_kn[MM], s_er[MM], s_ad[MM];
  __shared__ float s_wprev[NN], s_rprev[NN], s_prec[NN], s_usage[NN];
  __shared__ float s_w[NN];
  __shared__ float s_A[NN], s_B[NN];   // scratch: wg / w_addr / partials
  __shared__ float s_red[32];

  float* memb  = mem  + (size_t)b * NN * MM;
  float* linkb = link + (size_t)b * NN * NN;

  // ---- initialize mutable state from the (immutable) inputs ----
  {
    const float4* ms = (const float4*)(mem0  + (size_t)b * NN * MM);
    const float4* ls = (const float4*)(link0 + (size_t)b * NN * NN);
    float4* md = (float4*)memb;
    float4* ld = (float4*)linkb;
    for (int i = tid; i < NN*MM/4; i += 1024) md[i] = ms[i];
    for (int i = tid; i < NN*NN/4; i += 1024) ld[i] = ls[i];
  }
  s_wprev[tid] = ww0[b*NN + tid];
  s_rprev[tid] = rw0[b*NN + tid];
  s_prec [tid] = prec0[b*NN + tid];
  s_usage[tid] = usage0[b*NN + tid];
  __syncthreads();

  for (int t = 0; t < TT; ++t) {
    const int r = t*BB + b;
    for (int i = tid; i < RPL; i += 1024) s_rp[i] = rp_all[(size_t)r*RPL + i];
    for (int i = tid; i < WPL; i += 1024) s_wp[i] = wp_all[(size_t)r*WPL + i];
    __syncthreads();
    if (tid < MM) {
      s_er[tid] = sigmoidf(s_wp[262 + tid]);  // erase = sigmoid(p[M+3+K : 2M+3+K])
      s_ad[tid] = s_wp[518 + tid];            // add   = p[2M+3+K : 3M+3+K]
    }
    __syncthreads();

    // ================= WRITE HEAD =================
    addressing(s_wp, s_wprev, memb, s_kn, s_A, s_B, s_red);   // s_B = w_addr
    const float agate = sigmoidf(s_wp[774]);
    const float w = (1.f - agate) * s_B[tid];                 // alloc == 0 exactly
    s_w[tid] = w;
    wpub[(size_t)(t & 1)*BB*NN + (size_t)b*NN + tid] = w;     // publish for batch-sum

    gbar(bar, (unsigned)((t + 1) * gridDim.x));               // one grid sync per step

    float wsum = 0.f;
    #pragma unroll
    for (int bb = 0; bb < BB; ++bb)
      wsum += wpub[(size_t)(t & 1)*BB*NN + (size_t)bb*NN + tid];

    const float pn = s_prec[tid];

    // memory[n][:] = memory*(1 - w*erase) + w*add
    {
      float4* mrow = (float4*)(memb + (size_t)tid * MM);
      const float4* er4 = (const float4*)s_er;
      const float4* ad4 = (const float4*)s_ad;
      #pragma unroll 4
      for (int i = 0; i < MM/4; ++i) {
        float4 mv = mrow[i]; const float4 er = er4[i], ad = ad4[i];
        mv.x = mv.x*(1.f - w*er.x) + w*ad.x;
        mv.y = mv.y*(1.f - w*er.y) + w*ad.y;
        mv.z = mv.z*(1.f - w*er.z) + w*ad.z;
        mv.w = mv.w*(1.f - w*er.w) + w*ad.w;
        mrow[i] = mv;
      }
    }

    // link[n][k] = (1-w[n])*link[n][k] + prec[n]*w[k]; zero diagonal
    {
      float4* lrow = (float4*)(linkb + (size_t)tid * NN);
      const float4* w4 = (const float4*)s_w;
      const float omw = 1.f - w;
      for (int i = 0; i < NN/4; ++i) {
        if ((i & 31) == 0) __builtin_prefetch(lrow + i + 32, 0, 1);
        float4 L = lrow[i]; const float4 W = w4[i];
        L.x = omw*L.x + pn*W.x; L.y = omw*L.y + pn*W.y;
        L.z = omw*L.z + pn*W.z; L.w = omw*L.w + pn*W.w;
        lrow[i] = L;
      }
      linkb[(size_t)tid * NN + tid] = 0.f;
    }

    // precedence (cross-batch sum, per reference), usage, carry w
    s_prec [tid] = (1.f - wsum) * pn + w;
    s_usage[tid] = fminf(1.f, fmaxf(0.f, 0.99f * s_usage[tid] + w));
    s_wprev[tid] = w;
    __syncthreads();

    // ================= READ HEAD =================
    addressing(s_rp, s_rprev, memb, s_kn, s_A, s_B, s_red);   // s_B = wc (post-write memory)

    // fwd[n] = sum_k r[k]*link[k][n]   (coalesced column sweep)
    float fwd = 0.f;
    for (int k = 0; k < NN; ++k) {
      if ((k & 63) == 0) __builtin_prefetch(linkb + (size_t)(k + 64)*NN + tid, 0, 1);
      fwd += s_rprev[k] * linkb[(size_t)k * NN + tid];
    }
    // bwd[n] = sum_k r[k]*link[n][k]   (per-thread row stream)
    float bwd = 0.f;
    {
      const float4* lr  = (const float4*)(linkb + (size_t)tid * NN);
      const float4* rp4 = (const float4*)s_rprev;
      for (int i = 0; i < NN/4; ++i) {
        const float4 L = lr[i], R = rp4[i];
        bwd += L.x*R.x + L.y*R.y + L.z*R.z + L.w*R.w;
      }
    }
    // read-mode mixing pi = softmax(p[M+3+K : M+6+K])
    const float q0 = s_rp[262], q1 = s_rp[263], q2 = s_rp[264];
    const float qm = fmaxf(q0, fmaxf(q1, q2));
    const float f0 = expf(q0 - qm), f1 = expf(q1 - qm), f2 = expf(q2 - qm);
    const float qi = 1.f / (f0 + f1 + f2);
    const float wr = qi * (f0 * bwd + f1 * s_B[tid] + f2 * fwd);

    __syncthreads();            // all reads of s_rprev complete
    s_rprev[tid] = wr;
    __syncthreads();

    // reads[m] = sum_n wr[n] * memory[n][m]  (4-way split over n, coalesced in m)
    {
      const int m  = tid & 255;
      const int g2 = tid >> 8;
      float acc = 0.f;
      for (int n = g2; n < NN; n += 4)
        acc += s_rprev[n] * memb[(size_t)n * MM + m];
      s_A[tid] = acc;
      __syncthreads();
      if (tid < MM)
        out[((size_t)t*BB + b) * MM + tid] =
            s_A[tid] + s_A[256 + tid] + s_A[512 + tid] + s_A[768 + tid];
      __syncthreads();
    }
  }
}

// =====================================================================
extern "C" void kernel_launch(void* const* d_in, const int* in_sizes, int n_in,
                              void* d_out, int out_size, void* d_ws, size_t ws_size,
                              hipStream_t stream) {
  (void)in_sizes; (void)n_in; (void)out_size; (void)ws_size;

  const float* ctrl   = (const float*)d_in[0];
  const float* rW     = (const float*)d_in[1];
  const float* rb     = (const float*)d_in[2];
  const float* wW     = (const float*)d_in[3];
  const float* wb     = (const float*)d_in[4];
  const float* mem0   = (const float*)d_in[5];
  const float* link0  = (const float*)d_in[6];
  const float* prec0  = (const float*)d_in[7];
  const float* usage0 = (const float*)d_in[8];
  const float* rw0    = (const float*)d_in[9];
  const float* ww0    = (const float*)d_in[10];

  char* ws = (char*)d_ws;
  unsigned int* bar = (unsigned int*)(ws + OFF_BAR);
  float* wpub   = (float*)(ws + OFF_WPUB);
  float* rp_all = (float*)(ws + OFF_RP);
  float* wp_all = (float*)(ws + OFF_WP);
  float* memw   = (float*)(ws + OFF_MEM);
  float* linkw  = (float*)(ws + OFF_LINK);

  hipMemsetAsync(bar, 0, 256, stream);   // reset grid-barrier counter (capture-safe)

  ntm_proj<<<NTIL, 256, 0, stream>>>(ctrl, rW, rb, wW, wb, rp_all, wp_all);
  ntm_step<<<BB, 1024, 0, stream>>>(mem0, link0, prec0, usage0, rw0, ww0,
                                    rp_all, wp_all, memw, linkw, wpub, bar,
                                    (float*)d_out);
}
